// GridRnn_86715389706656
// MI455X (gfx1250) — matrix operand order repaired and checked by reading, not compiled, not run on hardware
//
#include <hip/hip_runtime.h>

// ---------------------------------------------------------------------------
// Problem constants: L,C,B,E,H,V,NC,NH = 4,8,2048,256,512,32000,32000,8
// ---------------------------------------------------------------------------
#define LL 4
#define CC 8
#define BB 2048
#define EE 256
#define HH 512
#define VV 32000
#define NCLS 32000
#define G3 1536            // 3*H
#define SB 16384           // S*B = C*B

typedef unsigned short u16;
typedef __attribute__((ext_vector_type(8)))  unsigned short u16x8;
typedef __attribute__((ext_vector_type(16))) __bf16        v16bf;
typedef __attribute__((ext_vector_type(8)))  float         v8f;

union FragBF { u16 s[16]; v16bf v; };

// Async global->LDS path (CDNA5): gate on builtin availability; fall back to
// the sync load/store staging otherwise.
#if defined(__HIP_DEVICE_COMPILE__) && __has_builtin(__builtin_amdgcn_global_load_async_to_lds_b128)
#define USE_ASYNC_LDS 1
#else
#define USE_ASYNC_LDS 0
#endif

#if USE_ASYNC_LDS
// Parameter type per hipcc diagnostic: pointer to 16-byte int vector,
// global (AS1) source and LDS (AS3) destination.
typedef int v4i_b128 __attribute__((vector_size(16)));
typedef __attribute__((address_space(1))) v4i_b128* gv4i_p;
typedef __attribute__((address_space(3))) v4i_b128* lv4i_p;
#endif

__device__ __forceinline__ u16 f2bf(float f) {
    unsigned u = __float_as_uint(f);
    unsigned r = u + 0x7FFFu + ((u >> 16) & 1u);   // round-to-nearest-even
    return (u16)(r >> 16);
}
__device__ __forceinline__ float bf2f(u16 h) {
    return __uint_as_float(((unsigned)h) << 16);
}

// ---------------------------------------------------------------------------
// f32 -> bf16 (grid-stride)
// ---------------------------------------------------------------------------
__global__ __launch_bounds__(256) void k_f32_to_bf16(const float* __restrict__ s,
                                                     u16* __restrict__ d, long n) {
    long i = (long)blockIdx.x * blockDim.x + threadIdx.x;
    long stride = (long)gridDim.x * blockDim.x;
    for (; i < n; i += stride) d[i] = f2bf(s[i]);
}

// ---------------------------------------------------------------------------
// f32 (batch,N,K) -> bf16 transposed (batch,K,N). One-time; writes coalesced.
// ---------------------------------------------------------------------------
__global__ __launch_bounds__(256) void k_convT(const float* __restrict__ s,
                                               u16* __restrict__ d,
                                               long total, int N, int K) {
    long i = (long)blockIdx.x * 256 + threadIdx.x;
    long stride = (long)gridDim.x * 256;
    long NK = (long)N * K;
    for (; i < total; i += stride) {
        long b = i / NK;
        long r = i - b * NK;
        int  k = (int)(r / N);
        int  n = (int)(r - (long)k * N);
        d[i] = f2bf(s[b * NK + (long)n * K + k]);
    }
}

// ---------------------------------------------------------------------------
// Layer-0 x part of xin (C,B,1024): cols [0,256) = embed[token] for cell 0,
// zeros for other cells. msg part written later by the layernorm kernel.
// ---------------------------------------------------------------------------
__global__ __launch_bounds__(256) void k_build_x0(const int* __restrict__ tok,
                                                  const float* __restrict__ embed,
                                                  u16* __restrict__ xin) {
    long idx = (long)blockIdx.x * 256 + threadIdx.x;     // over C*B*E
    int  e = (int)(idx & 255);
    long t = idx >> 8;
    int  b = (int)(t & 2047);
    int  c = (int)(t >> 11);
    float v = (c == 0) ? embed[(long)tok[b] * EE + e] : 0.0f;
    xin[((long)c * BB + b) * 1024 + e] = f2bf(v);
}

// ---------------------------------------------------------------------------
// Batched bf16 GEMM via v_wmma_f32_16x16x32_bf16.
//   D[m,n] = sum_k A[m,k] * Wt[k,n] + bias[n]
// A:  bf16 row-major, row stride lda.
// Wt: bf16 PRE-TRANSPOSED weights, (K,N) row-major, row stride N.
// Macro tile 128x128, BK=32, 256 threads = 8 waves.
// Wave w -> rows (w>>1)*32..+32, cols (w&1)*64..+64 (2x4 accumulators).
// Requires: M%128==0 (grid.y=M/128), N%128==0, K%32==0.
// ---------------------------------------------------------------------------
__global__ __launch_bounds__(256) void gemm_bf16_wmma(
    const u16* __restrict__ A, long sAb, int lda,
    const u16* __restrict__ Wt, long sBb,
    const float* __restrict__ bias, long sBias,
    void* __restrict__ D, long sDb, int ldd, int outBf16,
    int N, int K)
{
    __shared__ alignas(16) u16 As[128 * 40];   // 128 rows x 32 (+8 pad)
    __shared__ alignas(32) u16 Bs[32 * 144];   // 32 k-rows x 128 (+16 pad)

    const int tid  = threadIdx.x;
    const int wave = tid >> 5;
    const int lane = tid & 31;
    const int m0 = blockIdx.y * 128;
    const int n0 = blockIdx.x * 128;

    const u16*   Ab = A  + (long)blockIdx.z * sAb;
    const u16*   Bt = Wt + (long)blockIdx.z * sBb;
    const float* bp = bias + (long)blockIdx.z * sBias;

    const int wm = (wave >> 1) * 32;   // 0..96
    const int wn = (wave & 1) * 64;    // 0,64

    v8f acc[2][4] = {};

    // A staging map: 128 rows x 32 cols, 32B per thread
    const int ldr = tid >> 1;          // 0..127
    const int kc  = (tid & 1) * 16;    // 0 or 16
    // B staging map: 32 k-rows x 128 n-cols, 32B per thread
    const int bk  = tid >> 3;          // 0..31
    const int bnc = (tid & 7) * 16;    // 0..112

    for (int k0 = 0; k0 < K; k0 += 32) {
        const u16* srcA = Ab + (long)(m0 + ldr) * lda + k0 + kc;
        const u16* srcB = Bt + (long)(k0 + bk) * N + n0 + bnc;
#if USE_ASYNC_LDS
        {
            gv4i_p gA = (gv4i_p)(const void*)srcA;
            lv4i_p lA = (lv4i_p)(void*)&As[ldr * 40 + kc];
            gv4i_p gB = (gv4i_p)(const void*)srcB;
            lv4i_p lB = (lv4i_p)(void*)&Bs[bk * 144 + bnc];
            // imm offset applies to both global and LDS address (ISA 10.7):
            // second 16B chunk is +16B on both sides in our layout.
            __builtin_amdgcn_global_load_async_to_lds_b128(gA, lA, 0, 0);
            __builtin_amdgcn_global_load_async_to_lds_b128(gA, lA, 16, 0);
            __builtin_amdgcn_global_load_async_to_lds_b128(gB, lB, 0, 0);
            __builtin_amdgcn_global_load_async_to_lds_b128(gB, lB, 16, 0);
        }
#if __has_builtin(__builtin_amdgcn_s_wait_asynccnt)
        __builtin_amdgcn_s_wait_asynccnt(0);
#else
        asm volatile("s_wait_asynccnt 0" ::: "memory");
#endif
#else
        {
            u16x8 a0 = *(const u16x8*)srcA;
            u16x8 a1 = *(const u16x8*)(srcA + 8);
            *(u16x8*)&As[ldr * 40 + kc]     = a0;
            *(u16x8*)&As[ldr * 40 + kc + 8] = a1;
            u16x8 b0 = *(const u16x8*)srcB;
            u16x8 b1 = *(const u16x8*)(srcB + 8);
            *(u16x8*)&Bs[bk * 144 + bnc]     = b0;
            *(u16x8*)&Bs[bk * 144 + bnc + 8] = b1;
        }
#endif
        if (k0 + 32 < K) {             // global_prefetch_b8 for next K slab
            __builtin_prefetch(srcA + 32, 0, 1);
            __builtin_prefetch(srcB + (long)32 * N, 0, 1);
        }
        __syncthreads();

        // ---- fragments ----
        // A 16x32 bf16: lanes 0-15 -> M=lane, K {0..7,16..23};
        //               lanes 16-31 -> M=lane-16, K {8..15,24..31}.
        const int ml   = lane & 15;
        const int gsel = lane >> 4;
        FragBF afrag[2], bfrag[4];
        #pragma unroll
        for (int i = 0; i < 2; ++i) {
            const u16* p = &As[(wm + i * 16 + ml) * 40];
            u16x8 lo = *(const u16x8*)(p + 8 * gsel);
            u16x8 hi = *(const u16x8*)(p + 16 + 8 * gsel);
            #pragma unroll
            for (int e = 0; e < 8; ++e) { afrag[i].s[e] = lo[e]; afrag[i].s[8 + e] = hi[e]; }
        }
        // B 32x16: lanes 0-15 -> N=lane, K {0..7,16..23}; lanes 16-31 -> N=lane-16, K {8..15,24..31}.
        #pragma unroll
        for (int j = 0; j < 4; ++j) {
            const u16* p = &Bs[(8 * gsel) * 144 + wn + j * 16 + ml];
            #pragma unroll
            for (int e = 0; e < 8; ++e) { bfrag[j].s[e] = p[e * 144]; bfrag[j].s[8 + e] = p[(16 + e) * 144]; }
        }

        #pragma unroll
        for (int i = 0; i < 2; ++i)
            #pragma unroll
            for (int j = 0; j < 4; ++j)
                acc[i][j] = __builtin_amdgcn_wmma_f32_16x16x32_bf16(
                    false, afrag[i].v, false, bfrag[j].v,
                    (short)0, acc[i][j], false, false);
        __syncthreads();
    }

    // ---- epilogue: C/D layout lane = N + 16*(M>>3), vgpr r -> M = r + 8*(lane>>4) ----
    const int nl = lane & 15;
    const int mh = lane >> 4;
    #pragma unroll
    for (int i = 0; i < 2; ++i) {
        #pragma unroll
        for (int j = 0; j < 4; ++j) {
            #pragma unroll
            for (int r = 0; r < 8; ++r) {
                int row = m0 + wm + i * 16 + mh * 8 + r;
                int col = n0 + wn + j * 16 + nl;
                float v = acc[i][j][r] + bp[col];
                if (outBf16) {
                    u16* Dp = (u16*)D + (long)blockIdx.z * sDb;
                    Dp[(long)row * ldd + col] = f2bf(v);
                } else {
                    float* Dp = (float*)D + (long)blockIdx.z * sDb;
                    Dp[(long)row * ldd + col] = v;
                }
            }
        }
    }
}

// ---------------------------------------------------------------------------
// Attention core: one thread per (s, b, head). S=8, hd=64. qkv bf16 (S*B,1536).
// ---------------------------------------------------------------------------
__global__ __launch_bounds__(256) void k_attn_core(const u16* __restrict__ qkv,
                                                   u16* __restrict__ o) {
    int idx = blockIdx.x * 256 + threadIdx.x;   // < 8*2048*8
    int nh = idx & 7;
    int b  = (idx >> 3) & 2047;
    int s  = idx >> 14;

    const u16* qrow = qkv + ((long)s * BB + b) * G3 + nh * 64;
    float sc[8];
    float mx = -1e30f;
    #pragma unroll
    for (int t = 0; t < 8; ++t) {
        const u16* kr = qkv + ((long)t * BB + b) * G3 + 512 + nh * 64;
        float a = 0.0f;
        for (int d = 0; d < 64; ++d) a += bf2f(qrow[d]) * bf2f(kr[d]);
        sc[t] = a * 0.125f;                     // 1/sqrt(64)
        mx = fmaxf(mx, sc[t]);
    }
    float den = 0.0f;
    #pragma unroll
    for (int t = 0; t < 8; ++t) { sc[t] = __expf(sc[t] - mx); den += sc[t]; }
    float rden = 1.0f / den;

    u16* orow = o + ((long)s * BB + b) * HH + nh * 64;
    for (int d = 0; d < 64; ++d) {
        float a = 0.0f;
        #pragma unroll
        for (int t = 0; t < 8; ++t) {
            const u16* vr = qkv + ((long)t * BB + b) * G3 + 1024 + nh * 64;
            a += sc[t] * bf2f(vr[d]);
        }
        orow[d] = f2bf(a * rden);
    }
}

// ---------------------------------------------------------------------------
// LayerNorm over rows of (16384, 512) f32 -> bf16 msg into xin at msgoff.
// One wave32 per row; shuffle reduction.
// ---------------------------------------------------------------------------
__global__ __launch_bounds__(256) void k_attn_ln(const float* __restrict__ x,
                                                 const float* __restrict__ g,
                                                 const float* __restrict__ bta,
                                                 u16* __restrict__ xin, int msgoff) {
    int row  = blockIdx.x * 8 + (threadIdx.x >> 5);
    int lane = threadIdx.x & 31;
    const float* xr = x + (long)row * HH;
    float sum = 0.0f, sq = 0.0f;
    for (int i = lane; i < HH; i += 32) { float v = xr[i]; sum += v; sq += v * v; }
    #pragma unroll
    for (int o2 = 16; o2 > 0; o2 >>= 1) {
        sum += __shfl_xor(sum, o2, 32);
        sq  += __shfl_xor(sq,  o2, 32);
    }
    float mu  = sum * (1.0f / HH);
    float var = sq * (1.0f / HH) - mu * mu;
    float inv = rsqrtf(var + 1e-5f);
    u16* dst = xin + (long)row * 1024 + msgoff;
    for (int i = lane; i < HH; i += 32)
        dst[i] = f2bf((xr[i] - mu) * inv * g[i] + bta[i]);
}

// ---------------------------------------------------------------------------
// Fused GRU update over (C,B,H). Writes h_n (f32) into d_out; bf16 h_new into
// next layer's xin x-part; last layer's cell-0 slice as bf16 for head GEMM.
// ---------------------------------------------------------------------------
__global__ __launch_bounds__(256) void k_gru(const float* __restrict__ gi,
                                             const float* __restrict__ gh,
                                             const float* __restrict__ hin,
                                             float* __restrict__ hout,
                                             u16* __restrict__ xin_next,
                                             u16* __restrict__ hlast) {
    long idx = (long)blockIdx.x * 256 + threadIdx.x;   // < C*B*H
    int  j   = (int)(idx & 511);
    long row = idx >> 9;                               // c*B + b
    long base = row * G3;
    float ir = gi[base + j], iz = gi[base + 512 + j], inn = gi[base + 1024 + j];
    float hr = gh[base + j], hz = gh[base + 512 + j], hnn = gh[base + 1024 + j];
    float r = 1.0f / (1.0f + __expf(-(ir + hr)));
    float z = 1.0f / (1.0f + __expf(-(iz + hz)));
    float n = tanhf(inn + r * hnn);
    float hprev = hin[idx];
    float hnew = (1.0f - z) * n + z * hprev;
    hout[idx] = hnew;
    if (xin_next) xin_next[row * 1024 + j] = f2bf(hnew);
    if (hlast && row < BB) hlast[idx] = f2bf(hnew);    // c==0 rows only
}

// ---------------------------------------------------------------------------
// kernel_launch
// ---------------------------------------------------------------------------
extern "C" void kernel_launch(void* const* d_in, const int* in_sizes, int n_in,
                              void* d_out, int out_size, void* d_ws, size_t ws_size,
                              hipStream_t stream) {
    (void)in_sizes; (void)n_in; (void)out_size; (void)ws_size;

    const int*   tokens = (const int*)  d_in[0];
    const float* h      = (const float*)d_in[1];
    const float* embed  = (const float*)d_in[2];
    const float* W_ih0  = (const float*)d_in[3];
    const float* W_ihR  = (const float*)d_in[4];
    const float* W_hh   = (const float*)d_in[5];
    const float* b_ih   = (const float*)d_in[6];
    const float* b_hh   = (const float*)d_in[7];
    const float* awin   = (const float*)d_in[8];
    const float* abin   = (const float*)d_in[9];
    const float* awout  = (const float*)d_in[10];
    const float* about  = (const float*)d_in[11];
    const float* ln_g   = (const float*)d_in[12];
    const float* ln_b   = (const float*)d_in[13];
    const float* headw  = (const float*)d_in[14];
    const float* headb  = (const float*)d_in[15];

    float* y_out  = (float*)d_out;                       // (B, NC)
    float* hn_out = y_out + (size_t)BB * NCLS;           // (L, C, B, H)

    // ---- workspace carve-out ----
    char* ws = (char*)d_ws;
    size_t off = 0;
    auto take = [&](size_t bytes) -> char* {
        char* p = ws + off;
        off = (off + bytes + 255) & ~(size_t)255;
        return p;
    };
    const size_t nWih0 = (size_t)CC * G3 * (EE + HH);
    const size_t nWihR = (size_t)(LL - 1) * CC * G3 * (2 * HH);
    const size_t nWhh  = (size_t)LL * CC * G3 * HH;
    const size_t nWai  = (size_t)LL * G3 * HH;
    const size_t nWao  = (size_t)LL * HH * HH;
    const size_t nWhd  = (size_t)NCLS * HH;
    const size_t nCBH  = (size_t)CC * BB * HH;
    const size_t nCB3H = (size_t)CC * BB * G3;

    u16*   wih0_t = (u16*)take(nWih0 * 2);   // (C, 768, 1536)
    u16*   wihR_t = (u16*)take(nWihR * 2);   // (3*C, 1024, 1536)
    u16*   whh_t  = (u16*)take(nWhh * 2);    // (L*C, 512, 1536)
    u16*   wai_t  = (u16*)take(nWai * 2);    // (L, 512, 1536)
    u16*   wao_t  = (u16*)take(nWao * 2);    // (L, 512, 512)
    u16*   whd_t  = (u16*)take(nWhd * 2);    // (512, 32000)
    u16*   h_bf   = (u16*)take(nCBH * 2);
    u16*   qkv_bf = (u16*)take(nCB3H * 2);
    u16*   o_bf   = (u16*)take(nCBH * 2);
    float* attnout= (float*)take(nCBH * 4);
    u16*   xin_bf = (u16*)take((size_t)CC * BB * 1024 * 2);
    float* gi     = (float*)take(nCB3H * 4);
    float* gh     = (float*)take(nCB3H * 4);
    u16*   hlast  = (u16*)take((size_t)BB * HH * 2);

    auto conv = [&](const float* s, u16* d, size_t n) {
        int blocks = (int)((n + 255) / 256);
        k_f32_to_bf16<<<dim3(blocks), dim3(256), 0, stream>>>(s, d, (long)n);
    };
    auto convT = [&](const float* s, u16* d, size_t total, int N, int K) {
        int blocks = (int)((total + 255) / 256);
        k_convT<<<dim3(blocks), dim3(256), 0, stream>>>(s, d, (long)total, N, K);
    };
    auto gemm = [&](const u16* A, long sAb, int lda, const u16* Wt, long sBb,
                    const float* bias, long sBias, void* D, long sDb, int ldd,
                    int outBf16, int M, int N, int K, int batch) {
        dim3 g(N / 128, M / 128, batch), b(256);
        gemm_bf16_wmma<<<g, b, 0, stream>>>(A, sAb, lda, Wt, sBb, bias, sBias,
                                            D, sDb, ldd, outBf16, N, K);
    };

    // ---- one-time weight conversion (f32 -> bf16, transposed to (K,N)) ----
    convT(W_ih0, wih0_t, nWih0, G3, EE + HH);
    convT(W_ihR, wihR_t, nWihR, G3, 2 * HH);
    convT(W_hh,  whh_t,  nWhh,  G3, HH);
    convT(awin,  wai_t,  nWai,  G3, HH);
    convT(awout, wao_t,  nWao,  HH, HH);
    convT(headw, whd_t,  nWhd,  NCLS, HH);

    // ---- layer-0 x part ----
    k_build_x0<<<dim3((CC * BB * EE) / 256), dim3(256), 0, stream>>>(tokens, embed, xin_bf);

    for (int l = 0; l < LL; ++l) {
        // bf16 copy of this layer's input hidden state
        conv(h + (size_t)l * nCBH, h_bf, nCBH);

        // attention qkv: (M=16384, N=1536, K=512) -> bf16
        gemm(h_bf, 0, HH, wai_t + (size_t)l * G3 * HH, 0,
             abin + (size_t)l * G3, 0, qkv_bf, 0, G3, 1, SB, G3, HH, 1);

        // scores/softmax/weighted-sum -> o bf16
        k_attn_core<<<dim3((SB * 8) / 256), dim3(256), 0, stream>>>(qkv_bf, o_bf);

        // out proj: (M=16384, N=512, K=512) -> f32
        gemm(o_bf, 0, HH, wao_t + (size_t)l * HH * HH, 0,
             about + (size_t)l * HH, 0, attnout, 0, HH, 0, SB, HH, HH, 1);

        // layernorm -> msg (bf16) into xin at layer-dependent offset
        int msgoff = (l == 0) ? EE : HH;
        k_attn_ln<<<dim3(SB / 8), dim3(256), 0, stream>>>(
            attnout, ln_g + (size_t)l * HH, ln_b + (size_t)l * HH, xin_bf, msgoff);

        // gi = xin @ Wih^T + b_ih (per cell). K=768 (l=0) or 1024, lda=1024.
        int Kin = (l == 0) ? (EE + HH) : (2 * HH);
        const u16* Wl = (l == 0) ? wih0_t
                                 : wihR_t + (size_t)(l - 1) * CC * G3 * (2 * HH);
        gemm(xin_bf, (long)BB * 1024, 1024, Wl, (long)G3 * Kin,
             b_ih + (size_t)l * CC * G3, G3, gi, (long)BB * G3, G3, 0,
             BB, G3, Kin, CC);

        // gh = h @ Whh^T + b_hh (per cell). K=512.
        gemm(h_bf, (long)BB * HH, HH, whh_t + (size_t)l * CC * G3 * HH, (long)G3 * HH,
             b_hh + (size_t)l * CC * G3, G3, gh, (long)BB * G3, G3, 0,
             BB, G3, HH, CC);

        // GRU update
        k_gru<<<dim3((int)(nCBH / 256)), dim3(256), 0, stream>>>(
            gi, gh, h + (size_t)l * nCBH, hn_out + (size_t)l * nCBH,
            (l < LL - 1) ? xin_bf : nullptr,
            (l == LL - 1) ? hlast : nullptr);
    }

    // head: y = h_n[-1][0] @ head_w^T + head_b   (M=2048, N=32000, K=512)
    gemm(hlast, 0, HH, whd_t, 0, headb, 0, y_out, 0, NCLS, 0, BB, NCLS, HH, 1);
}
